// SimpleMLPMixer_27590869910207
// MI455X (gfx1250) — matrix operand order repaired and checked by reading, not compiled, not run on hardware
//
#include <hip/hip_runtime.h>

typedef __attribute__((ext_vector_type(16))) __bf16 v16bf;
typedef __attribute__((ext_vector_type(8)))  float  v8f;
typedef unsigned short u16;

union FragU { v16bf v; uint4 q[2]; };

__device__ inline u16 f2bf(float f) {
  unsigned int u = __float_as_uint(f);
  u += 0x7FFFu + ((u >> 16) & 1u);   // round-to-nearest-even
  return (u16)(u >> 16);
}
__device__ inline float geluf(float v) {
  return 0.5f * v * (1.0f + erff(v * 0.70710678118654752f));
}

// ---------------------------------------------------------------------------
// Generic batched bf16 GEMM:  C[z] = act( A[z](MxK) * Bt[z](NxK)^T + bias[z] )
// A row-major (M,K) bf16; Bt row-major (N,K) bf16 (pre-transposed weights).
// flags: bit0 = exact GELU, bit1 = bf16 output (else fp32).
// Block tile 128x128, K-step 32, 8 waves of 32 (wave32), wave tile 32x64.
// Double-buffered LDS: one barrier per K-step, global loads for tile k+1
// overlap with the 8 WMMAs of tile k.
// ---------------------------------------------------------------------------
#define BM 128
#define BN 128
#define BK 32
#define LDSP 40   // padded row length (elements) -> conflict-free b128 reads

__global__ __launch_bounds__(256)
void gemm_bf16(const u16* __restrict__ A, long long strideA,
               const u16* __restrict__ Bt, long long strideB,
               const float* __restrict__ bias, long long strideBias,
               void* __restrict__ Cv, long long strideC,
               int M, int N, int K, int ldc, int nstore, int flags)
{
  __shared__ u16 As[2][BM][LDSP];
  __shared__ u16 Bs[2][BN][LDSP];

  const int z = blockIdx.z;
  const u16* Ab = A + (size_t)z * (size_t)strideA;
  const u16* Bb = Bt + (size_t)z * (size_t)strideB;
  const float* biasb = bias ? (bias + (size_t)z * (size_t)strideBias) : nullptr;

  const int tid  = threadIdx.x;
  const int wave = tid >> 5;
  const int lane = tid & 31;
  const int laneLo = lane & 15;
  const int laneHi = lane >> 4;          // 0 or 1
  const int wy = wave >> 1;              // 0..3  (rows of 32)
  const int wx = wave & 1;               // 0..1  (cols of 64)

  const int rowBlock = blockIdx.y * BM;
  const int colBlock = blockIdx.x * BN;

  const int lrow = tid >> 1;             // 0..127
  const int lseg = (tid & 1) * 16;       // 0 or 16 (elements along K)

  const u16* ga = Ab + (size_t)(rowBlock + lrow) * (size_t)K + lseg;
  const u16* gb = Bb + (size_t)(colBlock + lrow) * (size_t)K + lseg;

  v8f acc[2][4];
#pragma unroll
  for (int i = 0; i < 2; ++i)
#pragma unroll
    for (int j = 0; j < 4; ++j)
      acc[i][j] = (v8f){0.f,0.f,0.f,0.f,0.f,0.f,0.f,0.f};

  // ---- prologue: stage tile 0 into LDS buffer 0 ----
  {
    uint4 ra0 = *(const uint4*)(ga);
    uint4 ra1 = *(const uint4*)(ga + 8);
    uint4 rb0 = *(const uint4*)(gb);
    uint4 rb1 = *(const uint4*)(gb + 8);
    *(uint4*)&As[0][lrow][lseg]     = ra0;
    *(uint4*)&As[0][lrow][lseg + 8] = ra1;
    *(uint4*)&Bs[0][lrow][lseg]     = rb0;
    *(uint4*)&Bs[0][lrow][lseg + 8] = rb1;
  }
  __syncthreads();

  int buf = 0;
  for (int k0 = 0; k0 < K; k0 += BK) {
    const bool more = (k0 + BK) < K;

    // issue next tile's global loads first (overlap with WMMAs below)
    uint4 ra0, ra1, rb0, rb1;
    if (more) {
      ra0 = *(const uint4*)(ga + (k0 + BK));
      ra1 = *(const uint4*)(ga + (k0 + BK) + 8);
      rb0 = *(const uint4*)(gb + (k0 + BK));
      rb1 = *(const uint4*)(gb + (k0 + BK) + 8);
    }

    // A fragments: lane<16 -> M=laneLo, K{0..7,16..23}; lane>=16 -> K{8..15,24..31}
    v16bf afrag[2], bfrag[4];
#pragma unroll
    for (int i = 0; i < 2; ++i) {
      const int r = wy * 32 + i * 16 + laneLo;
      FragU u;
      u.q[0] = *(const uint4*)&As[buf][r][laneHi * 8];
      u.q[1] = *(const uint4*)&As[buf][r][16 + laneHi * 8];
      afrag[i] = u.v;
    }
    // B fragments: lane -> column, lane<16 K{0..7,16..23}; lane>=16 K{8..15,24..31}
#pragma unroll
    for (int j = 0; j < 4; ++j) {
      const int c = wx * 64 + j * 16 + laneLo;
      FragU u;
      u.q[0] = *(const uint4*)&Bs[buf][c][laneHi * 8];
      u.q[1] = *(const uint4*)&Bs[buf][c][16 + laneHi * 8];
      bfrag[j] = u.v;
    }
#pragma unroll
    for (int i = 0; i < 2; ++i)
#pragma unroll
      for (int j = 0; j < 4; ++j)
        acc[i][j] = __builtin_amdgcn_wmma_f32_16x16x32_bf16(
            false, afrag[i], false, bfrag[j], (short)0, acc[i][j], false, false);

    if (more) {
      const int nb = buf ^ 1;
      *(uint4*)&As[nb][lrow][lseg]     = ra0;
      *(uint4*)&As[nb][lrow][lseg + 8] = ra1;
      *(uint4*)&Bs[nb][lrow][lseg]     = rb0;
      *(uint4*)&Bs[nb][lrow][lseg + 8] = rb1;
      __syncthreads();   // waits dscnt: stores visible, prior reads retired
      buf = nb;
    }
  }

  // Epilogue. C layout: VGPR r -> M = r + 8*laneHi, N = laneLo (per 16x16 tile)
  const bool do_gelu = (flags & 1) != 0;
  const bool out_bf  = (flags & 2) != 0;
  float* Cf = (float*)Cv;
  u16*   Cb = (u16*)Cv;
  const size_t cbase = (size_t)z * (size_t)strideC;
#pragma unroll
  for (int i = 0; i < 2; ++i) {
    const int row0 = rowBlock + wy * 32 + i * 16 + laneHi * 8;
#pragma unroll
    for (int j = 0; j < 4; ++j) {
      const int col = colBlock + wx * 64 + j * 16 + laneLo;
      if (col < nstore) {
        const float bv = biasb ? biasb[col] : 0.f;
#pragma unroll
        for (int r = 0; r < 8; ++r) {
          float v = acc[i][j][r] + bv;
          if (do_gelu) v = geluf(v);
          const size_t off = cbase + (size_t)(row0 + r) * (size_t)ldc + (size_t)col;
          if (out_bf) Cb[off] = f2bf(v); else Cf[off] = v;
        }
      }
    }
  }
}

// ---------------------------------------------------------------------------
// Weight convert + transpose: Wt[z](n,k) = bf16(W[z](k,n)), zero-padded.
// grid: (ceil(Kpad/256), Npad, z)
// ---------------------------------------------------------------------------
__global__ void convt_kernel(const float* __restrict__ W, u16* __restrict__ Wt,
                             int K, int N, int Kpad, long long sW, long long sWt)
{
  const int k = blockIdx.x * 256 + threadIdx.x;
  const int n = blockIdx.y;
  const int z = blockIdx.z;
  if (k >= Kpad) return;
  float v = (k < K && n < N) ? W[(size_t)z * (size_t)sW + (size_t)k * N + n] : 0.f;
  Wt[(size_t)z * (size_t)sWt + (size_t)n * Kpad + k] = f2bf(v);
}

// ---------------------------------------------------------------------------
// Input LN over (C,H,W)=12288 + ragged patch extraction + bf16 cast.
// xp layout: (16 patches, 2048 batch, 1216 padded) bf16. One block per sample.
// ---------------------------------------------------------------------------
__global__ void ln_patch_kernel(const float* __restrict__ x, const float* __restrict__ g,
                                const float* __restrict__ bta, u16* __restrict__ xp)
{
  const int bb = blockIdx.x;
  const int tid = threadIdx.x;
  const float* xb = x + (size_t)bb * 12288;
  float s = 0.f, s2 = 0.f;
  for (int i = tid; i < 12288; i += 256) { float v = xb[i]; s += v; s2 += v * v; }
  __shared__ float rs[256], rs2[256];
  rs[tid] = s; rs2[tid] = s2; __syncthreads();
  for (int off = 128; off > 0; off >>= 1) {
    if (tid < off) { rs[tid] += rs[tid + off]; rs2[tid] += rs2[tid + off]; }
    __syncthreads();
  }
  const float mean = rs[0] * (1.f / 12288.f);
  const float var  = rs2[0] * (1.f / 12288.f) - mean * mean;
  const float rstd = rsqrtf(var + 1e-5f);

  const int divs[5] = {0, 16, 36, 48, 64};
  const int wid[4]  = {16, 20, 12, 16};

  // zero tail padding of every patch row for this sample (disjoint from fill)
  for (int p = 0; p < 16; ++p) {
    const int sz = wid[p >> 2] * wid[p & 3] * 3;
    u16* dst = xp + ((size_t)p * 2048 + bb) * 1216;
    for (int idx = sz + tid; idx < 1216; idx += 256) dst[idx] = 0;
  }
  for (int i = tid; i < 12288; i += 256) {
    const int c = i >> 12;
    const int rem = i & 4095;
    const int h = rem >> 6, w = rem & 63;
    const int bi = (h < 16) ? 0 : (h < 36) ? 1 : (h < 48) ? 2 : 3;
    const int bj = (w < 16) ? 0 : (w < 36) ? 1 : (w < 48) ? 2 : 3;
    const int p = bi * 4 + bj;
    const int hh = wid[bi], wwd = wid[bj];
    const int idx = c * hh * wwd + (h - divs[bi]) * wwd + (w - divs[bj]);
    const float v = (xb[i] - mean) * rstd * g[i] + bta[i];
    xp[((size_t)p * 2048 + bb) * 1216 + idx] = f2bf(v);
  }
}

// ---------------------------------------------------------------------------
// Mixer LN: for (b,m) gather 512 vals from cat patch section, LN, write bf16
// xm layout (8, 2048, 512). 128 threads/block, block = b*8+m.
// ---------------------------------------------------------------------------
__global__ void mixer_ln_kernel(const float* __restrict__ cat, const float* __restrict__ g,
                                const float* __restrict__ bt, u16* __restrict__ xm)
{
  const int flat = blockIdx.x;
  const int m = flat & 7, bb = flat >> 3;
  const int tid = threadIdx.x;   // 128
  const float* cb = cat + (size_t)bb * 8192 + 4096 + m * 32;
  float vals[4]; float s = 0.f, s2 = 0.f;
#pragma unroll
  for (int t = 0; t < 4; ++t) {
    const int i = tid + t * 128;
    const float v = cb[(i >> 5) * 256 + (i & 31)];
    vals[t] = v; s += v; s2 += v * v;
  }
  __shared__ float rs[128], rs2[128];
  rs[tid] = s; rs2[tid] = s2; __syncthreads();
  for (int off = 64; off > 0; off >>= 1) {
    if (tid < off) { rs[tid] += rs[tid + off]; rs2[tid] += rs2[tid + off]; }
    __syncthreads();
  }
  const float mean = rs[0] * (1.f / 512.f);
  const float rstd = rsqrtf(rs2[0] * (1.f / 512.f) - mean * mean + 1e-5f);
  u16* dst = xm + ((size_t)m * 2048 + bb) * 512;
  const float* gm = g + m * 512;
  const float* bm = bt + m * 512;
#pragma unroll
  for (int t = 0; t < 4; ++t) {
    const int i = tid + t * 128;
    dst[i] = f2bf((vals[t] - mean) * rstd * gm[i] + bm[i]);
  }
}

// ---------------------------------------------------------------------------
// Row LN (fp32 in -> bf16 out), one block of 256 per row; n = 8192 or 2048.
// ---------------------------------------------------------------------------
__global__ void row_ln_kernel(const float* __restrict__ in, const float* __restrict__ g,
                              const float* __restrict__ bt, u16* __restrict__ out, int n)
{
  const int bb = blockIdx.x, tid = threadIdx.x;
  const float* ib = in + (size_t)bb * n;
  float s = 0.f, s2 = 0.f;
  for (int i = tid; i < n; i += 256) { float v = ib[i]; s += v; s2 += v * v; }
  __shared__ float rs[256], rs2[256];
  rs[tid] = s; rs2[tid] = s2; __syncthreads();
  for (int off = 128; off > 0; off >>= 1) {
    if (tid < off) { rs[tid] += rs[tid + off]; rs2[tid] += rs2[tid + off]; }
    __syncthreads();
  }
  const float inv = 1.f / (float)n;
  const float mean = rs[0] * inv;
  const float rstd = rsqrtf(rs2[0] * inv - mean * mean + 1e-5f);
  u16* ob = out + (size_t)bb * n;
  for (int i = tid; i < n; i += 256)
    ob[i] = f2bf((ib[i] - mean) * rstd * g[i] + bt[i]);
}

// ---------------------------------------------------------------------------
extern "C" void kernel_launch(void* const* d_in, const int* in_sizes, int n_in,
                              void* d_out, int out_size, void* d_ws, size_t ws_size,
                              hipStream_t stream)
{
  (void)in_sizes; (void)n_in; (void)out_size; (void)ws_size;
  const float* x    = (const float*)d_in[0];
  const float* lng  = (const float*)d_in[1];
  const float* lnb  = (const float*)d_in[2];
  const float* pW1  = (const float*)d_in[3];
  const float* pb1  = (const float*)d_in[4];
  const float* pW2  = (const float*)d_in[5];
  const float* pb2  = (const float*)d_in[6];
  const float* mlng = (const float*)d_in[7];
  const float* mlnb = (const float*)d_in[8];
  const float* mW1  = (const float*)d_in[9];
  const float* mb1  = (const float*)d_in[10];
  const float* mW2  = (const float*)d_in[11];
  const float* mb2  = (const float*)d_in[12];
  const float* clng = (const float*)d_in[13];
  const float* clnb = (const float*)d_in[14];
  const float* cW1  = (const float*)d_in[15];
  const float* cb1  = (const float*)d_in[16];
  const float* cW2  = (const float*)d_in[17];
  const float* cb2  = (const float*)d_in[18];
  const float* iW1  = (const float*)d_in[19];
  const float* ib1  = (const float*)d_in[20];
  const float* iW2  = (const float*)d_in[21];
  const float* ib2  = (const float*)d_in[22];
  const float* ilng = (const float*)d_in[23];
  const float* ilnb = (const float*)d_in[24];
  const float* clsW = (const float*)d_in[25];
  const float* clsb = (const float*)d_in[26];
  float* out = (float*)d_out;

  char* ws = (char*)d_ws;
  size_t off = 0;
  auto alloc = [&](size_t bytes) -> char* {
    char* r = ws + off;
    off += (bytes + 255) & ~(size_t)255;
    return r;
  };

  // bf16 transposed weights
  u16* pW1t  = (u16*)alloc((size_t)16 * 1216 * 256 * 2);
  u16* pW2t  = (u16*)alloc((size_t)16 * 256 * 256 * 2);
  u16* mW1t  = (u16*)alloc((size_t)8 * 512 * 512 * 2);
  u16* mW2t  = (u16*)alloc((size_t)8 * 512 * 512 * 2);
  u16* cW1t  = (u16*)alloc((size_t)2048 * 8192 * 2);
  u16* cW2t  = (u16*)alloc((size_t)2048 * 2048 * 2);
  u16* iW1t  = (u16*)alloc((size_t)2048 * 2048 * 2);
  u16* iW2t  = (u16*)alloc((size_t)2048 * 2048 * 2);
  u16* clsWt = (u16*)alloc((size_t)1024 * 2048 * 2);
  // activations (aliased regions)
  char* R1 = alloc((size_t)16 * 2048 * 1216 * 2);      // xp, later cat (67MB < 79.7MB)
  char* R2 = alloc((size_t)16 * 2048 * 256 * 2);       // hp, later first half of y0
  char* R3 = alloc((size_t)8 * 2048 * 512 * 2);        // xm, later second half of y0
  char* R4 = alloc((size_t)8 * 2048 * 512 * 2);        // hm, later h1 + ybf
  char* R5 = alloc((size_t)2048 * 2048 * 4);           // t (fp32)
  (void)R3;

  u16*   xp   = (u16*)R1;
  float* catf = (float*)R1;             // aliases xp (xp dead after GEMM1)
  u16*   hp   = (u16*)R2;
  u16*   xm   = (u16*)R3;
  u16*   hm   = (u16*)R4;
  u16*   y0   = (u16*)R2;               // 32MB spans R2+R3 (hp/xm dead by then)
  u16*   h1   = (u16*)R4;               // 8MB
  u16*   ybf  = (u16*)(R4 + (size_t)2048 * 2048 * 2);  // 8MB (hm dead by then)
  float* tbuf = (float*)R5;

  const dim3 blk(256);

  // ---- weight conversion + transpose (bf16) ----
  convt_kernel<<<dim3(5, 256, 16), blk, 0, stream>>>(pW1, pW1t, 1200, 256, 1216, 1200LL * 256, 1216LL * 256);
  convt_kernel<<<dim3(1, 256, 16), blk, 0, stream>>>(pW2, pW2t, 256, 256, 256, 256LL * 256, 256LL * 256);
  convt_kernel<<<dim3(2, 512, 8),  blk, 0, stream>>>(mW1, mW1t, 512, 512, 512, 512LL * 512, 512LL * 512);
  convt_kernel<<<dim3(2, 512, 8),  blk, 0, stream>>>(mW2, mW2t, 512, 512, 512, 512LL * 512, 512LL * 512);
  convt_kernel<<<dim3(32, 2048, 1), blk, 0, stream>>>(cW1, cW1t, 8192, 2048, 8192, 0, 0);
  convt_kernel<<<dim3(8, 2048, 1),  blk, 0, stream>>>(cW2, cW2t, 2048, 2048, 2048, 0, 0);
  convt_kernel<<<dim3(8, 2048, 1),  blk, 0, stream>>>(iW1, iW1t, 2048, 2048, 2048, 0, 0);
  convt_kernel<<<dim3(8, 2048, 1),  blk, 0, stream>>>(iW2, iW2t, 2048, 2048, 2048, 0, 0);
  convt_kernel<<<dim3(8, 1024, 1),  blk, 0, stream>>>(clsW, clsWt, 2048, 1000, 2048, 0, 0);

  // ---- input LN + ragged patch extraction ----
  ln_patch_kernel<<<dim3(2048), blk, 0, stream>>>(x, lng, lnb, xp);

  // ---- per-patch MLP ----
  gemm_bf16<<<dim3(2, 16, 16), blk, 0, stream>>>(xp, 2048LL * 1216, pW1t, 1216LL * 256,
      pb1, 256, (void*)hp, 2048LL * 256, 2048, 256, 1216, 256, 256, /*gelu|bf16*/3);
  gemm_bf16<<<dim3(2, 16, 16), blk, 0, stream>>>(hp, 2048LL * 256, pW2t, 256LL * 256,
      pb2, 256, (void*)(catf + 4096), 256, 2048, 256, 256, 8192, 256, /*fp32*/0);

  // ---- mixer channels ----
  mixer_ln_kernel<<<dim3(2048 * 8), dim3(128), 0, stream>>>(catf, mlng, mlnb, xm);
  gemm_bf16<<<dim3(4, 16, 8), blk, 0, stream>>>(xm, 2048LL * 512, mW1t, 512LL * 512,
      mb1, 512, (void*)hm, 2048LL * 512, 2048, 512, 512, 512, 512, 3);
  gemm_bf16<<<dim3(4, 16, 8), blk, 0, stream>>>(hm, 2048LL * 512, mW2t, 512LL * 512,
      mb2, 512, (void*)catf, 512, 2048, 512, 512, 8192, 512, 0);

  // ---- combiner ----
  row_ln_kernel<<<dim3(2048), blk, 0, stream>>>(catf, clng, clnb, y0, 8192);
  gemm_bf16<<<dim3(16, 16, 1), blk, 0, stream>>>(y0, 0, cW1t, 0, cb1, 0,
      (void*)h1, 0, 2048, 2048, 8192, 2048, 2048, 3);
  gemm_bf16<<<dim3(16, 16, 1), blk, 0, stream>>>(h1, 0, cW2t, 0, cb2, 0,
      (void*)ybf, 0, 2048, 2048, 2048, 2048, 2048, /*bf16 no gelu*/2);

  // ---- inner MLPs x4 (shared weights) ----
  for (int it = 0; it < 4; ++it) {
    gemm_bf16<<<dim3(16, 16, 1), blk, 0, stream>>>(ybf, 0, iW1t, 0, ib1, 0,
        (void*)h1, 0, 2048, 2048, 2048, 2048, 2048, 3);
    gemm_bf16<<<dim3(16, 16, 1), blk, 0, stream>>>(h1, 0, iW2t, 0, ib2, 0,
        (void*)tbuf, 0, 2048, 2048, 2048, 2048, 2048, 0);
    row_ln_kernel<<<dim3(2048), blk, 0, stream>>>(tbuf, ilng, ilnb, ybf, 2048);
  }

  // ---- classifier (N padded to 1024, store 1000 cols into d_out) ----
  gemm_bf16<<<dim3(8, 16, 1), blk, 0, stream>>>(ybf, 0, clsWt, 0, clsb, 0,
      (void*)out, 0, 2048, 1024, 2048, 1000, 1000, 0);
}
